// SwinStage_65953517797527
// MI455X (gfx1250) — matrix operand order repaired and checked by reading, not compiled, not run on hardware
//
#include <hip/hip_runtime.h>
#include <hip/hip_bf16.h>
#include <math.h>

// ---------------- problem constants ----------------
#define WSZ 8
#define GH 64
#define GW 64
#define BSZ 16
#define CDIM 256
#define NHEAD 8
#define HDIM 32
#define NTW 64            // tokens per window
#define MLPH 1024
#define CPBH 512
#define NLAY 2
#define NTOK (BSZ * GH * GW)   // 65536 token rows

typedef __attribute__((ext_vector_type(16))) _Float16 v16h;
typedef __attribute__((ext_vector_type(8)))  float    v8f;

union H16 { v16h v; _Float16 h[16]; };
union F8  { v8f  v; float    f[8];  };
union HP  { _Float16 h[2]; unsigned u; };

static __device__ inline v8f vzero8() {
  v8f z = {0.f, 0.f, 0.f, 0.f, 0.f, 0.f, 0.f, 0.f};
  return z;
}

static __device__ inline v8f wmma_f16(v16h a, v16h b, v8f c) {
  // D = A x B + C, 16x16x32, f16 in / f32 acc
  return __builtin_amdgcn_wmma_f32_16x16x32_f16(false, a, false, b, (short)0, c,
                                                false, false);
}

// A fragment: 16x32 (MxK), row-major source with row stride ld (halfs).
// lane L holds row r0+(L&15); halves j=0..15 -> K=k0+16*(L>>4)+j (contiguous).
static __device__ inline v16h frag_a(const _Float16* s, int ld, int r0, int k0) {
  int lane = threadIdx.x & 31;
  const _Float16* p = s + (size_t)(r0 + (lane & 15)) * ld + k0 + ((lane >> 4) << 4);
  H16 u;
#pragma unroll
  for (int j = 0; j < 16; ++j) u.h[j] = p[j];
  return u.v;
}

// B fragment: B is KxN; source is B^T (NxK) row-major with row stride ld.
// lane L holds col n0+(L&15); halves e=0..15 -> K=k0+16*(L>>4)+e (contiguous).
static __device__ inline v16h frag_b(const _Float16* s, int ld, int n0, int k0) {
  int lane = threadIdx.x & 31;
  const _Float16* p = s + (size_t)(n0 + (lane & 15)) * ld + k0 + ((lane >> 4) << 4);
  H16 u;
#pragma unroll
  for (int j = 0; j < 16; ++j) u.h[j] = p[j];
  return u.v;
}

// packed f32->f16 conversion of 64 contiguous elements into LDS (b32 stores)
static __device__ inline void cvt64_to_lds(const float* __restrict__ src,
                                           _Float16* __restrict__ dst) {
#pragma unroll 8
  for (int c = 0; c < 64; c += 2) {
    HP p;
    p.h[0] = (_Float16)src[c];
    p.h[1] = (_Float16)src[c + 1];
    *(unsigned*)(dst + c) = p.u;
  }
}

// ---------------- prep kernels ----------------

__global__ void cvt_f32_to_f16(const float* __restrict__ src,
                               _Float16* __restrict__ dst, int n) {
  int i = blockIdx.x * blockDim.x + threadIdx.x;
  int stride = gridDim.x * blockDim.x;
  for (; i < n; i += stride) dst[i] = (_Float16)src[i];
}

// continuous position bias: rel table -> MLP(2->512->NH) -> gather -> 16*sigmoid
// output biasTab[blk][h][m][n], (NLAY, NHEAD, 64, 64)
__global__ void cpb_bias_kernel(const float* __restrict__ cpb_w1,
                                const float* __restrict__ cpb_b1,
                                const float* __restrict__ cpb_w2,
                                float* __restrict__ biasTab) {
  int blk = blockIdx.x;
  __shared__ float tab[225 * NHEAD];
  const float* w1 = cpb_w1 + (size_t)blk * CPBH * 2;
  const float* b1 = cpb_b1 + (size_t)blk * CPBH;
  const float* w2 = cpb_w2 + (size_t)blk * NHEAD * CPBH;
  int t = threadIdx.x;
  if (t < 225) {
    int a = t / 15, b = t % 15;
    float xa = ((float)(a - 7)) / 7.f * 8.f;
    float xb = ((float)(b - 7)) / 7.f * 8.f;
    float va = copysignf(log2f(fabsf(xa) + 1.f) / 3.f, xa);
    float vb = copysignf(log2f(fabsf(xb) + 1.f) / 3.f, xb);
    float acc[NHEAD];
#pragma unroll
    for (int h = 0; h < NHEAD; ++h) acc[h] = 0.f;
    for (int j = 0; j < CPBH; ++j) {
      float hj = w1[j * 2] * va + w1[j * 2 + 1] * vb + b1[j];
      hj = fmaxf(hj, 0.f);
#pragma unroll
      for (int h = 0; h < NHEAD; ++h) acc[h] += w2[h * CPBH + j] * hj;
    }
#pragma unroll
    for (int h = 0; h < NHEAD; ++h) tab[t * NHEAD + h] = acc[h];
  }
  __syncthreads();
  for (int e = threadIdx.x; e < NHEAD * NTW * NTW; e += blockDim.x) {
    int h = e >> 12;
    int i = e & 4095;
    int t1 = i >> 6, t2 = i & 63;
    int d0 = (t1 >> 3) - (t2 >> 3) + 7;
    int d1 = (t1 & 7) - (t2 & 7) + 7;
    float v = tab[(d0 * 15 + d1) * NHEAD + h];
    biasTab[(size_t)blk * NHEAD * 4096 + e] = 16.f / (1.f + expf(-v));
  }
}

// ---------------- fused window attention (+post-LN +residual, in place) ----------
// one workgroup (8 waves) per window; 1024 workgroups per block
__global__ __launch_bounds__(256) void attn_kernel(
    float* __restrict__ xbuf,
    const _Float16* __restrict__ wqkv,  // (768,256) f16, row-major (out,in)
    const float* __restrict__ bqkv,     // (768,)
    const float* __restrict__ lscale,   // (NHEAD,)
    const float* __restrict__ biasTab,  // (NHEAD,64,64)
    const _Float16* __restrict__ wproj, // (256,256) f16
    const float* __restrict__ bproj,    // (256,)
    const float* __restrict__ n1s, const float* __restrict__ n1b,
    int shift) {
  __shared__ _Float16 xs[NTW * CDIM];      // 32 KB window tokens, f16
  __shared__ _Float16 os[NTW * CDIM];      // 32 KB concat head outputs
  __shared__ __align__(16) char scr[65536];// staged scratch / proj f32 overlay
  __shared__ float red_a[64 * 4];          // reduction partials
  __shared__ float red_b[64 * 4];
  __shared__ int rid[64];

  float*    qf = (float*)scr;                     // 64x32 f32
  float*    kf = (float*)(scr + 8192);            // 64x32 f32
  _Float16* qh = (_Float16*)(scr + 16384);        // 64x32 f16 (normalized)
  _Float16* kh = (_Float16*)(scr + 20480);        // 64x32 f16
  _Float16* vt = (_Float16*)(scr + 24576);        // 32x64 f16 (V^T, B-frag layout)
  float*    ss = (float*)(scr + 28672);           // 64x64 f32 scores
  _Float16* ps = (_Float16*)(scr + 45056);        // 64x64 f16 probs
  float*    pr = (float*)scr;                     // 64x256 f32 (after head loop)

  const int tid = threadIdx.x;
  const int wv = tid >> 5;
  const int bimg = blockIdx.x >> 6;
  const int widx = blockIdx.x & 63;
  const int wh = widx >> 3, ww = widx & 7;

  // token t (rolled coords) -> global row in xbuf
  auto grow = [&](int t) -> size_t {
    int rr = (wh * WSZ + (t >> 3) + shift) & 63;
    int cc = (ww * WSZ + (t & 7) + shift) & 63;
    return (size_t)bimg * 4096 + (size_t)rr * 64 + cc;
  };

  // load + convert window to LDS (4 threads per row, 64 cols each)
  {
    int row = tid >> 2, c0 = (tid & 3) * 64;
    cvt64_to_lds(xbuf + grow(row) * CDIM + c0, xs + row * CDIM + c0);
  }
  if (shift && tid < 64) {  // region ids for the shift mask (rolled coords)
    int R = wh * WSZ + (tid >> 3), Cc = ww * WSZ + (tid & 7);
    int rh = (R < GH - WSZ) ? 0 : ((R < GH - 4) ? 1 : 2);
    int rc = (Cc < GW - WSZ) ? 0 : ((Cc < GW - 4) ? 1 : 2);
    rid[tid] = rh * 3 + rc;
  }
  __syncthreads();

  for (int h = 0; h < NHEAD; ++h) {
    float scale = expf(fminf(lscale[h], 4.60517018598809f));  // ln(100)

    // ---- qkv for this head: 3 x (64x32) tiles, wave w -> tile (w>>1, w&1)
    {
      int mt = wv >> 1, nt = wv & 1;
#pragma unroll
      for (int p = 0; p < 3; ++p) {
        v8f acc = vzero8();
        const _Float16* wB = wqkv + (size_t)(p * CDIM + h * HDIM + nt * 16) * CDIM;
#pragma unroll
        for (int kk = 0; kk < 8; ++kk) {
          v16h a = frag_a(xs, CDIM, mt * 16, kk * 32);
          v16h b = frag_b(wB, CDIM, 0, kk * 32);
          acc = wmma_f16(a, b, acc);
        }
        float bias = bqkv[p * CDIM + h * HDIM + nt * 16 + (tid & 15)];
        F8 u; u.v = acc;
        int nloc = nt * 16 + (tid & 15);
        int mb = mt * 16 + ((tid >> 4) & 1) * 8;
#pragma unroll
        for (int e = 0; e < 8; ++e) {
          float v = u.f[e] + bias;
          if (p == 0)      qf[(mb + e) * HDIM + nloc] = v;
          else if (p == 1) kf[(mb + e) * HDIM + nloc] = v;
          else             vt[nloc * NTW + (mb + e)] = (_Float16)v;  // V^T
        }
      }
    }
    __syncthreads();

    // ---- cosine normalize q,k rows (128 row-tasks x 2 threads, 16 elems each)
    {
      int task = tid >> 1, half = tid & 1;   // task<64: q row; else k row
      const float* srcr = (task < 64) ? (qf + task * HDIM) : (kf + (task - 64) * HDIM);
      float s2 = 0.f;
#pragma unroll
      for (int j = 0; j < 16; ++j) { float v = srcr[half * 16 + j]; s2 += v * v; }
      red_a[task * 2 + half] = s2;
    }
    __syncthreads();
    {
      int task = tid >> 1, half = tid & 1;
      const float* srcr = (task < 64) ? (qf + task * HDIM) : (kf + (task - 64) * HDIM);
      _Float16* dstr = (task < 64) ? (qh + task * HDIM) : (kh + (task - 64) * HDIM);
      float s2 = red_a[task * 2] + red_a[task * 2 + 1];
      float inv = 1.f / (sqrtf(s2) + 1e-12f);
#pragma unroll
      for (int j = 0; j < 16; j += 2) {
        HP p;
        p.h[0] = (_Float16)(srcr[half * 16 + j] * inv);
        p.h[1] = (_Float16)(srcr[half * 16 + j + 1] * inv);
        *(unsigned*)(dstr + half * 16 + j) = p.u;
      }
    }
    __syncthreads();

    // ---- S = qh @ kh^T : 16 tiles of 16x16, K=32 (one wmma each)
#pragma unroll
    for (int tl = 0; tl < 2; ++tl) {
      int t = wv + tl * 8;
      int smt = t >> 2, snt = t & 3;
      v8f acc = vzero8();
      v16h a = frag_a(qh, HDIM, smt * 16, 0);
      v16h b = frag_b(kh, HDIM, snt * 16, 0);  // kh rows are B^T rows
      acc = wmma_f16(a, b, acc);
      F8 u; u.v = acc;
      int n = snt * 16 + (tid & 15);
      int mb = smt * 16 + ((tid >> 4) & 1) * 8;
#pragma unroll
      for (int e = 0; e < 8; ++e) {
        int m = mb + e;
        float v = u.f[e] * scale + biasTab[h * 4096 + m * 64 + n];
        if (shift && (rid[m] != rid[n])) v -= 100.f;
        ss[m * 64 + n] = v;
      }
    }
    __syncthreads();

    // ---- softmax rows (64 rows x 4 threads, 16 cols each) -> ps f16
    {
      int row = tid >> 2, seg = tid & 3;
      float* r = ss + row * 64 + seg * 16;
      float mx = -1e30f;
#pragma unroll
      for (int j = 0; j < 16; ++j) mx = fmaxf(mx, r[j]);
      red_a[tid] = mx;
    }
    __syncthreads();
    {
      int row = tid >> 2, seg = tid & 3;
      float* r = ss + row * 64 + seg * 16;
      float mx = fmaxf(fmaxf(red_a[row * 4], red_a[row * 4 + 1]),
                       fmaxf(red_a[row * 4 + 2], red_a[row * 4 + 3]));
      float sum = 0.f;
#pragma unroll
      for (int j = 0; j < 16; ++j) {
        float e = expf(r[j] - mx);
        r[j] = e;
        sum += e;
      }
      red_b[tid] = sum;
    }
    __syncthreads();
    {
      int row = tid >> 2, seg = tid & 3;
      const float* r = ss + row * 64 + seg * 16;
      float inv = 1.f / (red_b[row * 4] + red_b[row * 4 + 1] +
                         red_b[row * 4 + 2] + red_b[row * 4 + 3]);
      _Float16* d = ps + row * 64 + seg * 16;
#pragma unroll
      for (int j = 0; j < 16; j += 2) {
        HP p;
        p.h[0] = (_Float16)(r[j] * inv);
        p.h[1] = (_Float16)(r[j + 1] * inv);
        *(unsigned*)(d + j) = p.u;
      }
    }
    __syncthreads();

    // ---- O_h = P @ V : 8 tiles of 16x16, K=64 (V^T in B-frag layout)
    {
      int mt = wv >> 1, nt = wv & 1;
      v8f acc = vzero8();
#pragma unroll
      for (int kk = 0; kk < 2; ++kk) {
        v16h a = frag_a(ps, 64, mt * 16, kk * 32);
        v16h b = frag_b(vt, NTW, nt * 16, kk * 32);
        acc = wmma_f16(a, b, acc);
      }
      F8 u; u.v = acc;
      int n = h * HDIM + nt * 16 + (tid & 15);
      int mb = mt * 16 + ((tid >> 4) & 1) * 8;
#pragma unroll
      for (int e = 0; e < 8; ++e) os[(mb + e) * CDIM + n] = (_Float16)u.f[e];
    }
    __syncthreads();
  }  // heads

  // ---- projection: (64x256) @ (256x256)^T, 64 tiles, 8 per wave
#pragma unroll
  for (int j = 0; j < 8; ++j) {
    int t = wv * 8 + j;
    int mt = t >> 4, nt = t & 15;
    v8f acc = vzero8();
    const _Float16* wB = wproj + (size_t)(nt * 16) * CDIM;
#pragma unroll
    for (int kk = 0; kk < 8; ++kk) {
      v16h a = frag_a(os, CDIM, mt * 16, kk * 32);
      v16h b = frag_b(wB, CDIM, 0, kk * 32);
      acc = wmma_f16(a, b, acc);
    }
    float bias = bproj[nt * 16 + (tid & 15)];
    F8 u; u.v = acc;
    int n = nt * 16 + (tid & 15);
    int mb = mt * 16 + ((tid >> 4) & 1) * 8;
#pragma unroll
    for (int e = 0; e < 8; ++e) pr[(mb + e) * CDIM + n] = u.f[e] + bias;
  }
  __syncthreads();

  // ---- post-LN + residual (in place; window owns its token rows)
  {
    int row = tid >> 2, c0 = (tid & 3) * 64;
    const float* r = pr + row * CDIM + c0;
    float s = 0.f, s2 = 0.f;
#pragma unroll 8
    for (int j = 0; j < 64; ++j) { float v = r[j]; s += v; s2 += v * v; }
    red_a[tid] = s;
    red_b[tid] = s2;
  }
  __syncthreads();
  {
    int row = tid >> 2, c0 = (tid & 3) * 64;
    float s  = red_a[row * 4] + red_a[row * 4 + 1] + red_a[row * 4 + 2] + red_a[row * 4 + 3];
    float s2 = red_b[row * 4] + red_b[row * 4 + 1] + red_b[row * 4 + 2] + red_b[row * 4 + 3];
    float mean = s * (1.f / CDIM);
    float var = s2 * (1.f / CDIM) - mean * mean;
    float rstd = rsqrtf(var + 1e-5f);
    float* dst = xbuf + grow(row) * CDIM;
#pragma unroll 4
    for (int c = c0; c < c0 + 64; ++c) {
      float v = (pr[row * CDIM + c] - mean) * rstd * n1s[c] + n1b[c];
      dst[c] = dst[c] + v;
    }
  }
}

// ---------------- fused MLP (+post-LN +residual) --------------------------------
// one workgroup per 64 token rows; hidden split into 4 chunks of 256,
// second GEMM accumulated in registers across chunks.
__global__ __launch_bounds__(256) void mlp_kernel(
    const float* __restrict__ xin, float* __restrict__ xout,
    const _Float16* __restrict__ w1, const float* __restrict__ b1,
    const _Float16* __restrict__ w2, const float* __restrict__ b2,
    const float* __restrict__ n2s, const float* __restrict__ n2b) {
  __shared__ _Float16 xs[64 * CDIM];   // 32 KB
  __shared__ _Float16 hs[64 * CDIM];   // 32 KB hidden chunk (gelu'd, f16)
  __shared__ float pr[64 * CDIM];      // 64 KB output before LN
  __shared__ float red_a[64 * 4];
  __shared__ float red_b[64 * 4];

  const int tid = threadIdx.x;
  const int wv = tid >> 5;
  const size_t r0 = (size_t)blockIdx.x * 64;

  {
    int row = tid >> 2, c0 = (tid & 3) * 64;
    cvt64_to_lds(xin + (r0 + row) * CDIM + c0, xs + row * CDIM + c0);
  }
  __syncthreads();

  v8f oacc[8];
#pragma unroll
  for (int j = 0; j < 8; ++j) oacc[j] = vzero8();

  for (int hc = 0; hc < 4; ++hc) {
    // hidden chunk: (64x256) @ w1_chunk^T -> gelu -> hs
#pragma unroll
    for (int j = 0; j < 8; ++j) {
      int t = wv * 8 + j;
      int mt = t >> 4, nt = t & 15;
      v8f acc = vzero8();
      const _Float16* wB = w1 + (size_t)(hc * 256 + nt * 16) * CDIM;
#pragma unroll
      for (int kk = 0; kk < 8; ++kk) {
        v16h a = frag_a(xs, CDIM, mt * 16, kk * 32);
        v16h b = frag_b(wB, CDIM, 0, kk * 32);
        acc = wmma_f16(a, b, acc);
      }
      float bias = b1[hc * 256 + nt * 16 + (tid & 15)];
      F8 u; u.v = acc;
      int n = nt * 16 + (tid & 15);
      int mb = mt * 16 + ((tid >> 4) & 1) * 8;
#pragma unroll
      for (int e = 0; e < 8; ++e) {
        float x = u.f[e] + bias;
        float g = 0.5f * x * (1.f + erff(x * 0.70710678118654752f));  // exact gelu
        hs[(mb + e) * CDIM + n] = (_Float16)g;
      }
    }
    __syncthreads();

    // accumulate out += gelu(h_chunk) @ w2_chunk^T
#pragma unroll
    for (int j = 0; j < 8; ++j) {
      int t = wv * 8 + j;
      int mt = t >> 4, nt = t & 15;
      const _Float16* wB = w2 + (size_t)(nt * 16) * MLPH;
      v8f acc = oacc[j];
#pragma unroll
      for (int kk = 0; kk < 8; ++kk) {
        v16h a = frag_a(hs, CDIM, mt * 16, kk * 32);
        v16h b = frag_b(wB, MLPH, 0, hc * 256 + kk * 32);
        acc = wmma_f16(a, b, acc);
      }
      oacc[j] = acc;
    }
    __syncthreads();
  }

#pragma unroll
  for (int j = 0; j < 8; ++j) {
    int t = wv * 8 + j;
    int mt = t >> 4, nt = t & 15;
    float bias = b2[nt * 16 + (tid & 15)];
    F8 u; u.v = oacc[j];
    int n = nt * 16 + (tid & 15);
    int mb = mt * 16 + ((tid >> 4) & 1) * 8;
#pragma unroll
    for (int e = 0; e < 8; ++e) pr[(mb + e) * CDIM + n] = u.f[e] + bias;
  }
  __syncthreads();

  {
    int row = tid >> 2, c0 = (tid & 3) * 64;
    const float* r = pr + row * CDIM + c0;
    float s = 0.f, s2 = 0.f;
#pragma unroll 8
    for (int j = 0; j < 64; ++j) { float v = r[j]; s += v; s2 += v * v; }
    red_a[tid] = s;
    red_b[tid] = s2;
  }
  __syncthreads();
  {
    int row = tid >> 2, c0 = (tid & 3) * 64;
    float s  = red_a[row * 4] + red_a[row * 4 + 1] + red_a[row * 4 + 2] + red_a[row * 4 + 3];
    float s2 = red_b[row * 4] + red_b[row * 4 + 1] + red_b[row * 4 + 2] + red_b[row * 4 + 3];
    float mean = s * (1.f / CDIM);
    float var = s2 * (1.f / CDIM) - mean * mean;
    float rstd = rsqrtf(var + 1e-5f);
    const float* src = xin + (r0 + row) * CDIM;
    float* dst = xout + (r0 + row) * CDIM;
#pragma unroll 4
    for (int c = c0; c < c0 + 64; ++c)
      dst[c] = src[c] + (pr[row * CDIM + c] - mean) * rstd * n2s[c] + n2b[c];
  }
}

// ---------------- host launch ----------------------------------------------------
extern "C" void kernel_launch(void* const* d_in, const int* in_sizes, int n_in,
                              void* d_out, int out_size, void* d_ws, size_t ws_size,
                              hipStream_t stream) {
  const float* tokens  = (const float*)d_in[0];
  // d_in[1], d_in[2]: grid_h, grid_w (constant 64, hardcoded)
  const float* qkv_w   = (const float*)d_in[3];
  const float* qkv_b   = (const float*)d_in[4];
  const float* lscale  = (const float*)d_in[5];
  const float* cpb_w1  = (const float*)d_in[6];
  const float* cpb_b1  = (const float*)d_in[7];
  const float* cpb_w2  = (const float*)d_in[8];
  const float* proj_w  = (const float*)d_in[9];
  const float* proj_b  = (const float*)d_in[10];
  const float* norm1_s = (const float*)d_in[11];
  const float* norm1_b = (const float*)d_in[12];
  const float* mlp_w1  = (const float*)d_in[13];
  const float* mlp_b1  = (const float*)d_in[14];
  const float* mlp_w2  = (const float*)d_in[15];
  const float* mlp_b2  = (const float*)d_in[16];
  const float* norm2_s = (const float*)d_in[17];
  const float* norm2_b = (const float*)d_in[18];

  // workspace carve
  char* w = (char*)d_ws;
  float* xbuf = (float*)w;           w += (size_t)NTOK * CDIM * 4;       // 64 MB
  _Float16* qkv_w_h = (_Float16*)w;  w += (size_t)NLAY * 3 * CDIM * CDIM * 2;
  _Float16* proj_w_h = (_Float16*)w; w += (size_t)NLAY * CDIM * CDIM * 2;
  _Float16* w1h = (_Float16*)w;      w += (size_t)NLAY * MLPH * CDIM * 2;
  _Float16* w2h = (_Float16*)w;      w += (size_t)NLAY * CDIM * MLPH * 2;
  float* biasTab = (float*)w;        w += (size_t)NLAY * NHEAD * 4096 * 4;

  // x <- tokens (running activation buffer, f32)
  hipMemcpyAsync(xbuf, tokens, (size_t)NTOK * CDIM * sizeof(float),
                 hipMemcpyDeviceToDevice, stream);

  // convert weights to f16 once (halves L2 streaming width for WMMA B-frags)
  cvt_f32_to_f16<<<512, 256, 0, stream>>>(qkv_w, qkv_w_h, NLAY * 3 * CDIM * CDIM);
  cvt_f32_to_f16<<<512, 256, 0, stream>>>(proj_w, proj_w_h, NLAY * CDIM * CDIM);
  cvt_f32_to_f16<<<512, 256, 0, stream>>>(mlp_w1, w1h, NLAY * MLPH * CDIM);
  cvt_f32_to_f16<<<512, 256, 0, stream>>>(mlp_w2, w2h, NLAY * CDIM * MLPH);

  // continuous position bias tables
  cpb_bias_kernel<<<NLAY, 256, 0, stream>>>(cpb_w1, cpb_b1, cpb_w2, biasTab);

  const int nWinWG = BSZ * (GH / WSZ) * (GW / WSZ);  // 1024
  const int nRowWG = NTOK / 64;                      // 1024

  for (int i = 0; i < NLAY; ++i) {
    int shift = (i % 2 == 0) ? 0 : (WSZ / 2);
    attn_kernel<<<nWinWG, 256, 0, stream>>>(
        xbuf,
        qkv_w_h + (size_t)i * 3 * CDIM * CDIM, qkv_b + (size_t)i * 3 * CDIM,
        lscale + (size_t)i * NHEAD, biasTab + (size_t)i * NHEAD * 4096,
        proj_w_h + (size_t)i * CDIM * CDIM, proj_b + (size_t)i * CDIM,
        norm1_s + (size_t)i * CDIM, norm1_b + (size_t)i * CDIM, shift);

    float* out = (i == NLAY - 1) ? (float*)d_out : xbuf;
    mlp_kernel<<<nRowWG, 256, 0, stream>>>(
        xbuf, out,
        w1h + (size_t)i * MLPH * CDIM, mlp_b1 + (size_t)i * MLPH,
        w2h + (size_t)i * CDIM * MLPH, mlp_b2 + (size_t)i * CDIM,
        norm2_s + (size_t)i * CDIM, norm2_b + (size_t)i * CDIM);
  }
}
